// FactorizedMoEExperts_64587718197840
// MI455X (gfx1250) — compile-verified, matrix operand order checked by reading, not run on hardware
//
#include <hip/hip_runtime.h>
#include <hip/hip_bf16.h>

// ---------------------------------------------------------------------------
// FactorizedMoEExperts for MI455X (gfx1250, wave32, WMMA + TDM).
// Compute-bound (~326 GFLOP vs ~15us of HBM traffic at 23.3 TB/s): all GEMMs
// use V_WMMA_F32_16X16X32_BF16 (K=32/issue), f32 accumulation. fp32 operands
// are converted on the fly (v_cvt_pk_bf16_f32) while bf16 operand tiles are
// DMA'd into LDS by the Tensor Data Mover (TENSOR_LOAD_TO_LDS + TENSORcnt),
// keeping the vector pipes free for WMMA.
// Workspace requirement: ~105 MB (w: 256KB, lr bf16: 8MB, h bf16: 96MB).
// ---------------------------------------------------------------------------

#define DEV __device__ __forceinline__

constexpr int NE   = 32;    // experts
constexpr int NG   = 4;     // shared-Vh groups
constexpr int PER  = NE / NG;
constexpr int DM   = 2048;  // d_model
constexpr int RR   = 512;   // low-rank dim
constexpr int FF   = 768;   // expert intermediate
constexpr int NT   = 2048;  // tokens
constexpr int TOPK = 8;

constexpr int BM = 64;          // M tile (all GEMMs)
constexpr int BNW = 128;        // wide N tile (low-rank / down kernels)
constexpr int BN  = 64;         // narrow N tile (gate/up kernel)
constexpr int BK  = 32;         // K per stage == WMMA K
constexpr int LDSTR = BK + 8;   // 40 elems = 80B row stride (16B aligned)
constexpr int NTHREADS = 128;   // 4 waves of 32

typedef __attribute__((ext_vector_type(16))) __bf16 v16bf;
typedef __attribute__((ext_vector_type(8)))  float  v8f;
typedef __attribute__((ext_vector_type(4)))  unsigned int v4u;
typedef __attribute__((ext_vector_type(8)))  int v8i;
typedef __attribute__((ext_vector_type(4)))  int v4i;
typedef __attribute__((address_space(3)))    unsigned short lds_u16;

struct Pack32 { uint4 a, b; };   // 32 bytes == sizeof(v16bf)

#if __has_builtin(__builtin_amdgcn_tensor_load_to_lds) && \
    __has_builtin(__builtin_amdgcn_s_wait_tensorcnt)
#define HAVE_TDM 1
#else
#define HAVE_TDM 0
#endif

// Hardware convert: lowers to v_cvt_pk_bf16_f32.
DEV void cvt_store4(unsigned short* dst, float4 v) {
    __bf16* d = (__bf16*)dst;
    d[0] = (__bf16)v.x;
    d[1] = (__bf16)v.y;
    d[2] = (__bf16)v.z;
    d[3] = (__bf16)v.w;
}

// Stage a ROWS x BK fp32 tile from global -> bf16 LDS tile (stride LDSTR).
template <int ROWS>
DEV void stage_f32(unsigned short* lds, const float* __restrict__ g, size_t ldg) {
    const int tid = threadIdx.x;
    if (ROWS == 64) {                       // 16 elems / thread
        const int row = tid >> 1;
        const int col = (tid & 1) * 16;
        const float4* src = (const float4*)(g + (size_t)row * ldg + col);
        unsigned short* dst = lds + row * LDSTR + col;
#pragma unroll
        for (int i = 0; i < 4; ++i) cvt_store4(dst + 4 * i, src[i]);
    } else {                                // ROWS == 128: full row / thread
        const int row = tid;
        const float4* src = (const float4*)(g + (size_t)row * ldg);
        unsigned short* dst = lds + row * LDSTR;
#pragma unroll
        for (int i = 0; i < 8; ++i) cvt_store4(dst + 4 * i, src[i]);
    }
}

// Fallback staging for bf16 tiles (used only if the TDM builtin is absent).
DEV void stage_bf16_64(unsigned short* lds, const unsigned short* __restrict__ g, size_t ldg) {
    const int tid = threadIdx.x;
    const int row = tid >> 1;
    const int col = (tid & 1) * 16;
    const uint4* src = (const uint4*)(g + (size_t)row * ldg + col);
    uint4 v0 = src[0];
    uint4 v1 = src[1];
    *(uint4*)(lds + row * LDSTR + col)     = v0;
    *(uint4*)(lds + row * LDSTR + col + 8) = v1;
}

#if HAVE_TDM
// Tensor Data Mover: DMA a (rows x BK) bf16 tile from global into LDS,
// reproducing the LDSTR-padded layout via the D# pad fields:
//   row = 32 elems * 2B = 64B = 16 DWORDs  -> pad_interval = 3 (16 DWORDs)
//   pad = 16B = 4 DWORDs                   -> pad_amount   = 3 (4 DWORDs)
// D# encoding per CDNA5 ISA.
DEV void tdm_load_bf16_tile(unsigned int lds_addr, const unsigned short* gsrc,
                            int rows, int ld_elems) {
    unsigned long long ga = (unsigned long long)(const void*)gsrc;
    v4u g0;
    g0[0] = 1u;                                    // count=1, user descriptor
    g0[1] = lds_addr;                              // lds_addr [63:32]
    g0[2] = (unsigned)(ga & 0xFFFFFFFFu);          // global_addr lo
    g0[3] = (unsigned)((ga >> 32) & 0x01FFFFFFu)   // global_addr hi (57b)
          | (2u << 30);                            // type = 2 ("image")
    v8i g1;
    g1[0] = (1 << 16)                              // data_size = 2 bytes
          | (1 << 20)                              // pad_enable
          | (3 << 22)                              // pad_interval: 16 DWORDs
          | (3 << 25);                             // pad_amount:   4 DWORDs
    g1[1] = (BK & 0xFFFF) << 16;                   // tensor_dim0 lo16 @ [63:48]
    g1[2] = ((rows & 0xFFFF) << 16);               // tensor_dim0 hi / dim1 lo
    g1[3] = (BK << 16);                            // tensor_dim1 hi / tile_dim0
    g1[4] = rows & 0xFFFF;                         // tile_dim1 (tile_dim2 = 0)
    g1[5] = ld_elems;                              // tensor_dim0_stride lo32
    g1[6] = 0;
    g1[7] = 0;
    v4i g2 = {0, 0, 0, 0};                         // 2D tensor: groups 2/3 unused
    v4i g3 = {0, 0, 0, 0};
#if defined(__clang_major__) && __clang_major__ >= 23
    v8i g4 = {0, 0, 0, 0, 0, 0, 0, 0};
    __builtin_amdgcn_tensor_load_to_lds(g0, g1, g2, g3, g4, 0);
#else
    __builtin_amdgcn_tensor_load_to_lds(g0, g1, g2, g3, 0);
#endif
}
#endif

DEV unsigned int lds_addr_of(const unsigned short* p) {
    return (unsigned int)(unsigned long long)(lds_u16*)p;
}

// Load one 16x32 bf16 fragment from an LDS tile starting at row row0.
// ISA layout (CDNA5 ISA §7.12.2, 16-bit A 16x32): lane L -> row M=L&15,
// VGPR0..3 = K{hi*8..hi*8+7}, VGPR4..7 = K{16+hi*8..}, hi = L>>4.
// B tiles stored as [N][K] use the identical mapping (M <-> N symmetric).
DEV v16bf load_frag(const unsigned short* lds, int row0) {
    const int lane = threadIdx.x & 31;
    const unsigned short* p = lds + (row0 + (lane & 15)) * LDSTR + ((lane >> 4) * 8);
    uint4 c0 = *(const uint4*)p;          // K chunk 0
    uint4 c1 = *(const uint4*)(p + 16);   // K chunk 1 (+16 elems)
    Pack32 t{c0, c1};
    return __builtin_bit_cast(v16bf, t);
}

DEV v8f wmma_bf16(v16bf a, v16bf b, v8f c) {
    return __builtin_amdgcn_wmma_f32_16x16x32_bf16(
        /*neg_a=*/false, a, /*neg_b=*/false, b,
        /*c_mod=*/(short)0, c, /*reuse_a=*/false, /*reuse_b=*/false);
}

DEV void zero_acc(v8f& c) {
#pragma unroll
    for (int v = 0; v < 8; ++v) c[v] = 0.0f;
}

// C/D layout: m = 8*(lane>>4) + vgpr, n = lane&15.
DEV void store_bf16_frag(unsigned short* base, size_t ld, int m0, int n0, v8f c) {
    const int lane = threadIdx.x & 31;
    const int n = lane & 15, mh = (lane >> 4) * 8;
    __bf16* b = (__bf16*)base;
#pragma unroll
    for (int v = 0; v < 8; ++v)
        b[(size_t)(m0 + mh + v) * ld + (n0 + n)] = (__bf16)c[v];
}

DEV void store_f32_frag(float* base, size_t ld, int m0, int n0, v8f c) {
    const int lane = threadIdx.x & 31;
    const int n = lane & 15, mh = (lane >> 4) * 8;
#pragma unroll
    for (int v = 0; v < 8; ++v)
        base[(size_t)(m0 + mh + v) * ld + (n0 + n)] = c[v];
}

// ---------------------------------------------------------------------------
// Kernel 1a/1b: combine-weight matrix w[E,T] (matches reference index_add).
// ---------------------------------------------------------------------------
__global__ void k_zero_w(float* __restrict__ w) {
    int i = blockIdx.x * blockDim.x + threadIdx.x;
    if (i < NE * NT) w[i] = 0.0f;
}

__global__ void k_scatter_w(const long long* __restrict__ idx,
                            const float* __restrict__ tw,
                            float* __restrict__ w) {
    int i = blockIdx.x * blockDim.x + threadIdx.x;
    if (i < NT * TOPK) {
        int t = i / TOPK;
        int e = (int)idx[i];
        atomicAdd(&w[(size_t)e * NT + t], tw[i]);   // global_atomic_add_f32
    }
}

// ---------------------------------------------------------------------------
// Kernel 2: lr[g] = hidden @ v_hat[g]^T   (M=T, N=R, K=D), output bf16.
// Block tile 64x128, wave tile 32x64 -> 8 WMMAs per 12 b128 fragment loads.
// (Both operands fp32 -> staged through the cvt path; no TDM here.)
// ---------------------------------------------------------------------------
__global__ __launch_bounds__(NTHREADS) void k_lowrank(
    const float* __restrict__ hidden, const float* __restrict__ vhat,
    unsigned short* __restrict__ lr) {
    __shared__ __align__(16) unsigned short As[BM  * LDSTR];
    __shared__ __align__(16) unsigned short Bs[BNW * LDSTR];

    const int g  = blockIdx.z;
    const int m0 = blockIdx.y * BM;
    const int n0 = blockIdx.x * BNW;
    const float* A = hidden + (size_t)m0 * DM;
    const float* B = vhat + ((size_t)g * RR + n0) * DM;

    const int wave = (int)(threadIdx.x >> 5);
    const int wm = (wave >> 1) * 32, wn = (wave & 1) * 64;

    v8f acc[2][4];
#pragma unroll
    for (int i = 0; i < 2; ++i)
#pragma unroll
        for (int j = 0; j < 4; ++j) zero_acc(acc[i][j]);

    for (int kt = 0; kt < DM; kt += BK) {
        stage_f32<BM >(As, A + kt, DM);
        stage_f32<BNW>(Bs, B + kt, DM);
        __syncthreads();
        v16bf a[2], b[4];
#pragma unroll
        for (int i = 0; i < 2; ++i) a[i] = load_frag(As, wm + i * 16);
#pragma unroll
        for (int j = 0; j < 4; ++j) b[j] = load_frag(Bs, wn + j * 16);
#pragma unroll
        for (int i = 0; i < 2; ++i)
#pragma unroll
            for (int j = 0; j < 4; ++j)
                acc[i][j] = wmma_bf16(a[i], b[j], acc[i][j]);
        __syncthreads();
    }

    unsigned short* C = lr + ((size_t)g * NT + m0) * RR + n0;
#pragma unroll
    for (int i = 0; i < 2; ++i)
#pragma unroll
        for (int j = 0; j < 4; ++j)
            store_bf16_frag(C, RR, wm + i * 16, wn + j * 16, acc[i][j]);
}

// ---------------------------------------------------------------------------
// Kernel 3: per expert, fused gate/up GEMM + SiLU + routing weight.
// A tile (bf16 lr) is DMA'd by the TDM; gate AND up weight tiles are staged
// via cvt so the epilogue has both halves locally.
// h[e,t,f] = silu(gate)*up*w[e,t]  (bf16).
// ---------------------------------------------------------------------------
__global__ __launch_bounds__(NTHREADS) void k_gateup(
    const unsigned short* __restrict__ lr, const float* __restrict__ ugu,
    const float* __restrict__ w, unsigned short* __restrict__ h) {
    __shared__ __align__(16) unsigned short As[BM * LDSTR];
    __shared__ __align__(16) unsigned short Bg[BN * LDSTR];
    __shared__ __align__(16) unsigned short Bu[BN * LDSTR];

    const int e  = blockIdx.z;
    const int g  = e / PER;
    const int m0 = blockIdx.y * BM;
    const int n0 = blockIdx.x * BN;              // within [0, FF)
    const unsigned short* A = lr + ((size_t)g * NT + m0) * RR;
    const float* BGp = ugu + ((size_t)e * (2 * FF) + n0) * RR;
    const float* BUp = BGp + (size_t)FF * RR;

    const int wave = (int)(threadIdx.x >> 5);
    const int wm = (wave >> 1) * 32, wn = (wave & 1) * 32;
    const unsigned int as_lds = lds_addr_of(As);

    v8f cg[2][2], cu[2][2];
#pragma unroll
    for (int i = 0; i < 2; ++i)
#pragma unroll
        for (int j = 0; j < 2; ++j) { zero_acc(cg[i][j]); zero_acc(cu[i][j]); }

    for (int kt = 0; kt < RR; kt += BK) {
#if HAVE_TDM
        if (wave == 0) tdm_load_bf16_tile(as_lds, A + kt, BM, RR);
#else
        stage_bf16_64(As, A + kt, RR);
#endif
        stage_f32<BN>(Bg, BGp + kt, RR);
        stage_f32<BN>(Bu, BUp + kt, RR);
#if HAVE_TDM
        if (wave == 0) __builtin_amdgcn_s_wait_tensorcnt(0);
#endif
        __syncthreads();
        v16bf a0  = load_frag(As, wm);
        v16bf a1  = load_frag(As, wm + 16);
        v16bf bg0 = load_frag(Bg, wn);
        v16bf bg1 = load_frag(Bg, wn + 16);
        v16bf bu0 = load_frag(Bu, wn);
        v16bf bu1 = load_frag(Bu, wn + 16);
        cg[0][0] = wmma_bf16(a0, bg0, cg[0][0]);
        cg[0][1] = wmma_bf16(a0, bg1, cg[0][1]);
        cg[1][0] = wmma_bf16(a1, bg0, cg[1][0]);
        cg[1][1] = wmma_bf16(a1, bg1, cg[1][1]);
        cu[0][0] = wmma_bf16(a0, bu0, cu[0][0]);
        cu[0][1] = wmma_bf16(a0, bu1, cu[0][1]);
        cu[1][0] = wmma_bf16(a1, bu0, cu[1][0]);
        cu[1][1] = wmma_bf16(a1, bu1, cu[1][1]);
        __syncthreads();
    }

    const int lane = threadIdx.x & 31;
    const int n = lane & 15, mh = (lane >> 4) * 8;
    __bf16* hb = (__bf16*)h;
#pragma unroll
    for (int i = 0; i < 2; ++i) {
        float wrow[8];
#pragma unroll
        for (int v = 0; v < 8; ++v)
            wrow[v] = w[(size_t)e * NT + m0 + wm + i * 16 + mh + v];
#pragma unroll
        for (int j = 0; j < 2; ++j) {
#pragma unroll
            for (int v = 0; v < 8; ++v) {
                float gate = cg[i][j][v];
                float up   = cu[i][j][v];
                float s    = gate / (1.0f + __expf(-gate));   // silu
                float val  = s * up * wrow[v];
                size_t t = (size_t)(m0 + wm + i * 16 + mh + v);
                size_t f = (size_t)(n0 + wn + j * 16 + n);
                hb[((size_t)e * NT + t) * FF + f] = (__bf16)val;
            }
        }
    }
}

// ---------------------------------------------------------------------------
// Kernel 4: out[t,d] = sum_e h[e] @ down[e]^T. Block owns a 64x128 (t,d)
// tile and loops experts x K -> no atomics, deterministic accumulation.
// A tile (bf16 h) via TDM; down_proj streamed through cvt staging with an
// L2 prefetch of the next expert's tiles (global_prefetch_b8).
// ---------------------------------------------------------------------------
__global__ __launch_bounds__(NTHREADS) void k_down(
    const unsigned short* __restrict__ h, const float* __restrict__ down,
    float* __restrict__ out) {
    __shared__ __align__(16) unsigned short As[BM  * LDSTR];
    __shared__ __align__(16) unsigned short Bs[BNW * LDSTR];

    const int m0 = blockIdx.y * BM;
    const int n0 = blockIdx.x * BNW;             // within [0, D)

    const int wave = (int)(threadIdx.x >> 5);
    const int wm = (wave >> 1) * 32, wn = (wave & 1) * 64;
    const unsigned int as_lds = lds_addr_of(As);

    v8f acc[2][4];
#pragma unroll
    for (int i = 0; i < 2; ++i)
#pragma unroll
        for (int j = 0; j < 4; ++j) zero_acc(acc[i][j]);

    for (int e = 0; e < NE; ++e) {
        const unsigned short* A = h + ((size_t)e * NT + m0) * FF;
        const float* B = down + ((size_t)e * DM + n0) * FF;
        if (e + 1 < NE) {   // warm L2 for the next expert's streamed weights
            __builtin_prefetch(down + ((size_t)(e + 1) * DM + n0) * FF, 0, 2);
            __builtin_prefetch(h + ((size_t)(e + 1) * NT + m0) * FF, 0, 2);
        }
        for (int kt = 0; kt < FF; kt += BK) {
#if HAVE_TDM
            if (wave == 0) tdm_load_bf16_tile(as_lds, A + kt, BM, FF);
#else
            stage_bf16_64(As, A + kt, FF);
#endif
            stage_f32<BNW>(Bs, B + kt, FF);
#if HAVE_TDM
            if (wave == 0) __builtin_amdgcn_s_wait_tensorcnt(0);
#endif
            __syncthreads();
            v16bf a[2], b[4];
#pragma unroll
            for (int i = 0; i < 2; ++i) a[i] = load_frag(As, wm + i * 16);
#pragma unroll
            for (int j = 0; j < 4; ++j) b[j] = load_frag(Bs, wn + j * 16);
#pragma unroll
            for (int i = 0; i < 2; ++i)
#pragma unroll
                for (int j = 0; j < 4; ++j)
                    acc[i][j] = wmma_bf16(a[i], b[j], acc[i][j]);
            __syncthreads();
        }
    }

    float* C = out + (size_t)m0 * DM + n0;
#pragma unroll
    for (int i = 0; i < 2; ++i)
#pragma unroll
        for (int j = 0; j < 4; ++j)
            store_f32_frag(C, DM, wm + i * 16, wn + j * 16, acc[i][j]);
}

// ---------------------------------------------------------------------------
extern "C" void kernel_launch(void* const* d_in, const int* in_sizes, int n_in,
                              void* d_out, int out_size, void* d_ws, size_t ws_size,
                              hipStream_t stream) {
    (void)in_sizes; (void)n_in; (void)out_size; (void)ws_size;

    const float*     hidden = (const float*)d_in[0];
    const long long* tk_idx = (const long long*)d_in[1];   // int64
    const float*     tk_w   = (const float*)d_in[2];
    const float*     vhat   = (const float*)d_in[3];
    const float*     ugu    = (const float*)d_in[4];
    const float*     down   = (const float*)d_in[5];
    float*           out    = (float*)d_out;

    // Workspace layout (bytes): w | lr(bf16) | h(bf16)
    const size_t W_BYTES  = (size_t)NE * NT * sizeof(float);          // 256 KB
    const size_t LR_BYTES = (size_t)NG * NT * RR * sizeof(short);     // 8 MB
    char* ws = (char*)d_ws;
    float*          w_buf  = (float*)ws;
    unsigned short* lr_buf = (unsigned short*)(ws + W_BYTES);
    unsigned short* h_buf  = (unsigned short*)(ws + W_BYTES + LR_BYTES);

    // 1) routing weights
    k_zero_w<<<(NE * NT + 255) / 256, 256, 0, stream>>>(w_buf);
    k_scatter_w<<<(NT * TOPK + 255) / 256, 256, 0, stream>>>(tk_idx, tk_w, w_buf);

    // 2) shared low-rank projection: [G, T, R]
    k_lowrank<<<dim3(RR / BNW, NT / BM, NG), NTHREADS, 0, stream>>>(hidden, vhat, lr_buf);

    // 3) per-expert fused gate/up + SiLU + routing weight: [E, T, FF]
    k_gateup<<<dim3(FF / BN, NT / BM, NE), NTHREADS, 0, stream>>>(lr_buf, ugu, w_buf, h_buf);

    // 4) down projection with cross-expert accumulation: [T, D]
    k_down<<<dim3(DM / BNW, NT / BM, 1), NTHREADS, 0, stream>>>(h_buf, down, out);
}
